// BasedBlock_13503377178738
// MI455X (gfx1250) — matrix operand order repaired and checked by reading, not compiled, not run on hardware
//
#include <hip/hip_runtime.h>
#include <cstdint>

typedef __bf16 bf16;
typedef __attribute__((ext_vector_type(16))) __bf16 v16bf;
typedef __attribute__((ext_vector_type(8)))  float  v8f;

constexpr int kB  = 4;
constexpr int kL  = 2048;
constexpr int kD  = 1024;
constexpr int kH  = 16;
constexpr int kF  = 16;
constexpr int kHD = 64;
constexpr int kDD = 273;   // 1 + f + f*f
constexpr int kDP = 288;   // padded to 9*32 for K-steps of 32

static __device__ __forceinline__ v8f wmma_bf16(v16bf a, v16bf b, v8f c) {
  return __builtin_amdgcn_wmma_f32_16x16x32_bf16(false, a, false, b, (short)0, c, false, false);
}

// Issue per-lane 16B global->LDS async copy (tracked by ASYNCcnt).
static __device__ __forceinline__ void async_copy_b128(void* lds_ptr, const void* gptr) {
  unsigned l = (unsigned)(uintptr_t)lds_ptr;     // low 32 bits of generic = LDS offset
  asm volatile("global_load_async_to_lds_b128 %0, %1, off"
               :: "v"(l), "v"(gptr) : "memory");
}
static __device__ __forceinline__ void async_wait0() {
  asm volatile("s_wait_asynccnt 0" ::: "memory");
}

// Taylor feature map value for padded dim index dI, given the 16 raw values.
// phi[0]=1 ; phi[1+i]=x_i*0.5 ; phi[17+16i+j]=x_i*x_j/(sqrt2*sqrt(f))
static __device__ __forceinline__ float featv(const float* q, int dI) {
  if (dI == 0) return 1.0f;
  if (dI < 1 + kF) return q[dI - 1] * 0.5f;
  if (dI < kDD) { int t = dI - 17; return q[t >> 4] * q[t & 15] * 0.1767766952966369f; }
  return 0.0f;
}

// ---------------------------------------------------------------------------
// GEMM: C[M,N] = A[M,K] @ W[K,N] + bias[N]   (f32 in/out, bf16 WMMA compute)
// block = 256 threads (8 waves), tile 128x128, K-step 32.
// Double-buffered: A tile streamed with GLOBAL_LOAD_ASYNC_TO_LDS_B128,
// W tile register-prefetched then converted after WMMAs issue.
// ---------------------------------------------------------------------------
__global__ __launch_bounds__(256) void gemm_bias_wmma(
    const float* __restrict__ A, const float* __restrict__ W,
    const float* __restrict__ bias, float* __restrict__ C,
    int M, int N, int K)
{
  __shared__ float As32[2][128 * 32];   // raw f32 A tiles (async target)
  __shared__ bf16  Bs[2][128 * 32];     // transposed bf16 W tiles: Bs[n][k]

  const int tid  = threadIdx.x;
  const int wv   = tid >> 5;
  const int lane = tid & 31;
  const int r    = lane & 15;
  const int hf   = lane >> 4;
  const int m0   = blockIdx.y * 128;
  const int n0   = blockIdx.x * 128;
  const int wr   = wv & 3;    // wave row (4)  -> 32 rows each
  const int wc   = wv >> 2;   // wave col (2)  -> 64 cols each

  const v8f vz = {0.f,0.f,0.f,0.f,0.f,0.f,0.f,0.f};
  v8f acc[2][4];
#pragma unroll
  for (int i = 0; i < 2; ++i)
#pragma unroll
    for (int j = 0; j < 4; ++j) acc[i][j] = vz;

  // per-thread copy coordinates
  int arow[4], akq[4], bkk[4], bnq[4];
#pragma unroll
  for (int i = 0; i < 4; ++i) {
    int lin = tid + i * 256;
    arow[i] = lin >> 3; akq[i] = (lin & 7) << 2;   // A: 128 rows x 8 chunks
    bkk[i] = lin >> 5; bnq[i] = (lin & 31) << 2;   // W: 32 k-rows x 32 n-chunks
  }

  // ---- prologue: tile 0 ----
#pragma unroll
  for (int i = 0; i < 4; ++i)
    async_copy_b128(&As32[0][arow[i] * 32 + akq[i]],
                    A + (size_t)(m0 + arow[i]) * K + akq[i]);
  {
    float4 bt[4];
#pragma unroll
    for (int i = 0; i < 4; ++i)
      bt[i] = *(const float4*)(W + (size_t)bkk[i] * N + n0 + bnq[i]);
#pragma unroll
    for (int i = 0; i < 4; ++i) {
      Bs[0][(bnq[i] + 0) * 32 + bkk[i]] = (bf16)bt[i].x;
      Bs[0][(bnq[i] + 1) * 32 + bkk[i]] = (bf16)bt[i].y;
      Bs[0][(bnq[i] + 2) * 32 + bkk[i]] = (bf16)bt[i].z;
      Bs[0][(bnq[i] + 3) * 32 + bkk[i]] = (bf16)bt[i].w;
    }
  }
  async_wait0();
  __syncthreads();

  for (int k0 = 0; k0 < K; k0 += 32) {
    const int bu = (k0 >> 5) & 1;
    const int nb = bu ^ 1;
    const bool has_next = (k0 + 32 < K);

    // issue next-tile traffic first (async A copy + W register prefetch)
    float4 bt[4];
    if (has_next) {
#pragma unroll
      for (int i = 0; i < 4; ++i)
        async_copy_b128(&As32[nb][arow[i] * 32 + akq[i]],
                        A + (size_t)(m0 + arow[i]) * K + (k0 + 32) + akq[i]);
#pragma unroll
      for (int i = 0; i < 4; ++i)
        bt[i] = *(const float4*)(W + (size_t)(k0 + 32 + bkk[i]) * N + n0 + bnq[i]);
    }

    // compute on current buffer
    v16bf af[2], bf_[4];
#pragma unroll
    for (int mt = 0; mt < 2; ++mt) {
      int row = wr * 32 + mt * 16 + r;
#pragma unroll
      for (int e = 0; e < 8; ++e) {
        af[mt][e]     = (bf16)As32[bu][row * 32 + 16 * hf + e];
        af[mt][8 + e] = (bf16)As32[bu][row * 32 + 16 * hf + 8 + e];
      }
    }
#pragma unroll
    for (int nt = 0; nt < 4; ++nt) {
      int n = wc * 64 + nt * 16 + r;
#pragma unroll
      for (int e = 0; e < 16; ++e) bf_[nt][e] = Bs[bu][n * 32 + 16 * hf + e];
    }
#pragma unroll
    for (int mt = 0; mt < 2; ++mt)
#pragma unroll
      for (int nt = 0; nt < 4; ++nt)
        acc[mt][nt] = wmma_bf16(af[mt], bf_[nt], acc[mt][nt]);

    // store prefetched W tile into the next buffer (after WMMAs issue)
    if (has_next) {
#pragma unroll
      for (int i = 0; i < 4; ++i) {
        Bs[nb][(bnq[i] + 0) * 32 + bkk[i]] = (bf16)bt[i].x;
        Bs[nb][(bnq[i] + 1) * 32 + bkk[i]] = (bf16)bt[i].y;
        Bs[nb][(bnq[i] + 2) * 32 + bkk[i]] = (bf16)bt[i].z;
        Bs[nb][(bnq[i] + 3) * 32 + bkk[i]] = (bf16)bt[i].w;
      }
    }
    async_wait0();
    __syncthreads();
  }

#pragma unroll
  for (int mt = 0; mt < 2; ++mt)
#pragma unroll
    for (int nt = 0; nt < 4; ++nt) {
      int col = n0 + wc * 64 + nt * 16 + r;
      float bsv = bias[col];
#pragma unroll
      for (int r8 = 0; r8 < 8; ++r8) {
        int row = m0 + wr * 32 + mt * 16 + r8 + 8 * hf;
        C[(size_t)row * N + col] = acc[mt][nt][r8] + bsv;
      }
    }
}

// ---------------------------------------------------------------------------
// Chunked linear-attention scan. One block per (b,h), 256 threads, chunk T=16.
// State KV (64 x 288, transposed [hd][Dp]) lives in LDS as f32 + bf16 mirror.
// ---------------------------------------------------------------------------
constexpr int S_KV   = 0;                         // float 64*288
constexpr int S_KVB  = S_KV  + 64 * kDP * 4;      // bf16  64*288
constexpr int S_QF   = S_KVB + 64 * kDP * 2;      // bf16  16*288
constexpr int S_KF   = S_QF  + 16 * kDP * 2;      // bf16  16*288
constexpr int S_KFT  = S_KF  + 16 * kDP * 2;      // bf16  288*32 (K-padded)
constexpr int S_VT   = S_KFT + kDP * 32 * 2;      // bf16  64*32  (K-padded)
constexpr int S_SAB  = S_VT  + 64 * 32 * 2;       // bf16  16*32  (masked A)
constexpr int S_SA   = S_SAB + 16 * 32 * 2;       // float 16*16
constexpr int S_DEN  = S_SA  + 16 * 16 * 4;       // float 16
constexpr int S_QK2  = S_DEN + 16 * 4;            // float 32
constexpr int S_KST  = S_QK2 + 32 * 4;            // float 288
constexpr int S_Q16  = S_KST + kDP * 4;           // float 256
constexpr int S_K16  = S_Q16 + 256 * 4;           // float 256
constexpr int SCAN_LDS_SIZE = S_K16 + 256 * 4;    // ~157 KB (needs CDNA5 320KB LDS)

__global__ __launch_bounds__(256) void linear_scan_wmma(
    const float* __restrict__ qlin, const float* __restrict__ klin,
    const float* __restrict__ vbuf, const float* __restrict__ kv0,
    const float* __restrict__ kst0, float* __restrict__ lin_out,
    float* __restrict__ kv_out, float* __restrict__ kst_out)
{
  extern __shared__ char smem[];
  float* KV   = (float*)(smem + S_KV);
  bf16*  KVb  = (bf16*)(smem + S_KVB);
  bf16*  sQf  = (bf16*)(smem + S_QF);
  bf16*  sKf  = (bf16*)(smem + S_KF);
  bf16*  sKfT = (bf16*)(smem + S_KFT);
  bf16*  sVt  = (bf16*)(smem + S_VT);
  bf16*  sSab = (bf16*)(smem + S_SAB);
  float* sSa  = (float*)(smem + S_SA);
  float* sDen = (float*)(smem + S_DEN);
  float* qk2  = (float*)(smem + S_QK2);
  float* kst  = (float*)(smem + S_KST);
  float* q16  = (float*)(smem + S_Q16);
  float* k16  = (float*)(smem + S_K16);

  const int tid  = threadIdx.x;
  const int wv   = tid >> 5;
  const int lane = tid & 31;
  const int r    = lane & 15;
  const int hf   = lane >> 4;
  const int bh   = blockIdx.x;           // b*16 + h
  const int bb   = bh >> 4;
  const int hh   = bh & 15;
  const v8f vz = {0.f,0.f,0.f,0.f,0.f,0.f,0.f,0.f};

  // ---- init state + zero K-pad regions (t=16..31 stays zero forever) ----
  for (int j = tid; j < 64 * kDP; j += 256) {
    int c = j / kDP, dI = j - c * kDP;
    float v = (dI < kDD) ? kv0[((size_t)bh * kDD + dI) * 64 + c] : 0.f;
    KV[c * kDP + dI] = v;
    KVb[c * kDP + dI] = (bf16)v;
  }
  for (int dI = tid; dI < kDP; dI += 256)
    kst[dI] = (dI < kDD) ? kst0[(size_t)bh * kDD + dI] : 0.f;
  for (int j = tid; j < kDP * 32; j += 256) sKfT[j] = (bf16)0.f;
  for (int j = tid; j < 64 * 32; j += 256)  sVt[j]  = (bf16)0.f;
  for (int j = tid; j < 16 * 32; j += 256)  sSab[j] = (bf16)0.f;
  __syncthreads();

  for (int c0 = 0; c0 < kL; c0 += 16) {
    // ---- stage raw q,k (16x16 each) ----
    {
      int t = tid >> 4, i = tid & 15;
      q16[tid] = qlin[((size_t)(bb * kL + c0 + t)) * (kH * kF) + hh * kF + i];
      k16[tid] = klin[((size_t)(bb * kL + c0 + t)) * (kH * kF) + hh * kF + i];
    }
    __syncthreads();

    // ---- feature maps Qf,Kf (16x288 bf16), KfT (288x32 bf16), Vt (64x32) ----
    for (int j = tid; j < 16 * kDP; j += 256) {
      int t = j / kDP, dI = j - t * kDP;
      float qv = featv(q16 + t * 16, dI);
      float kv = featv(k16 + t * 16, dI);
      sQf[t * kDP + dI] = (bf16)qv;
      sKf[t * kDP + dI] = (bf16)kv;
      sKfT[dI * 32 + t] = (bf16)kv;
    }
    for (int j = tid; j < 64 * 16; j += 256) {
      int c = j >> 4, t = j & 15;
      sVt[c * 32 + t] =
          (bf16)vbuf[((size_t)(bb * kL + c0 + t)) * kD + hh * kHD + c];
    }
    __syncthreads();

    // ---- stage 2: inter-chunk numerator / intra A / q·kstate ----
    v8f accN = vz;  // persists across barriers for waves 0..3
    if (wv < 4) {
      int n = wv * 16 + r;
      for (int kb = 0; kb < 9; ++kb) {
        v16bf a, b;
#pragma unroll
        for (int e = 0; e < 8; ++e) {
          a[e]     = sQf[r * kDP + kb * 32 + 16 * hf + e];
          a[8 + e] = sQf[r * kDP + kb * 32 + 16 * hf + 8 + e];
        }
#pragma unroll
        for (int e = 0; e < 16; ++e) b[e] = KVb[n * kDP + kb * 32 + 16 * hf + e];
        accN = wmma_bf16(a, b, accN);
      }
    } else if (wv == 4) {
      v8f accA = vz;
      for (int kb = 0; kb < 9; ++kb) {
        v16bf a, b;
#pragma unroll
        for (int e = 0; e < 8; ++e) {
          a[e]     = sQf[r * kDP + kb * 32 + 16 * hf + e];
          a[8 + e] = sQf[r * kDP + kb * 32 + 16 * hf + 8 + e];
        }
#pragma unroll
        for (int e = 0; e < 16; ++e) b[e] = sKf[r * kDP + kb * 32 + 16 * hf + e];
        accA = wmma_bf16(a, b, accA);
      }
#pragma unroll
      for (int r8 = 0; r8 < 8; ++r8) sSa[(r8 + 8 * hf) * 16 + r] = accA[r8];
    } else if (wv == 5) {
      int t = lane >> 1, hsel = lane & 1;
      float s = 0.f;
      for (int dI = hsel * 144; dI < hsel * 144 + 144; ++dI)
        s += featv(q16 + t * 16, dI) * kst[dI];
      qk2[lane] = s;
    }
    __syncthreads();

    // ---- stage 3: causal mask + cumulative denominator ----
    if (tid < 16) {
      float run = 0.f;
      for (int s2 = 0; s2 <= tid; ++s2) run += sSa[tid * 16 + s2];
      for (int s2 = 0; s2 < 16; ++s2)
        sSab[tid * 32 + s2] = (bf16)((s2 <= tid) ? sSa[tid * 16 + s2] : 0.f);
      sDen[tid] = qk2[2 * tid] + qk2[2 * tid + 1] + run + 1e-6f;
    }
    __syncthreads();

    // ---- stage 4: num += maskedA @ V ; write normalized output ----
    if (wv < 4) {
      v16bf a, b;
#pragma unroll
      for (int e = 0; e < 8; ++e) {
        a[e]     = sSab[r * 32 + 16 * hf + e];
        a[8 + e] = sSab[r * 32 + 16 * hf + 8 + e];
      }
      int n = wv * 16 + r;
#pragma unroll
      for (int e = 0; e < 16; ++e) b[e] = sVt[n * 32 + 16 * hf + e];
      accN = wmma_bf16(a, b, accN);
#pragma unroll
      for (int r8 = 0; r8 < 8; ++r8) {
        int t = r8 + 8 * hf;
        lin_out[((size_t)(bb * kL + c0 + t)) * kD + hh * kHD + wv * 16 + r] =
            accN[r8] / sDen[t];
      }
    }

    // ---- stage 5: state update KV += Kf^T(288x16) @ V(16x64); 72 tiles/8 waves
    for (int ii = 0; ii < 9; ++ii) {
      int tile = wv * 9 + ii;
      int mt = tile >> 2, nt = tile & 3;
      v16bf a, b;
      int m = mt * 16 + r;
#pragma unroll
      for (int e = 0; e < 8; ++e) {
        a[e]     = sKfT[m * 32 + 16 * hf + e];
        a[8 + e] = sKfT[m * 32 + 16 * hf + 8 + e];
      }
      int n = nt * 16 + r;
#pragma unroll
      for (int e = 0; e < 16; ++e) b[e] = sVt[n * 32 + 16 * hf + e];
      v8f cfr;
#pragma unroll
      for (int r8 = 0; r8 < 8; ++r8)
        cfr[r8] = KV[n * kDP + mt * 16 + r8 + 8 * hf];
      v8f dd = wmma_bf16(a, b, cfr);
#pragma unroll
      for (int r8 = 0; r8 < 8; ++r8) {
        int mi = mt * 16 + r8 + 8 * hf;
        KV[n * kDP + mi]  = dd[r8];
        KVb[n * kDP + mi] = (bf16)dd[r8];
      }
    }
    // kstate += column sums of Kf (recomputed in f32)
    for (int dI = tid; dI < kDP; dI += 256) {
      float s = 0.f;
      for (int t = 0; t < 16; ++t) s += featv(k16 + t * 16, dI);
      kst[dI] += s;
    }
    __syncthreads();
  }

  // ---- emit final states ----
  for (int j = tid; j < kDD * 64; j += 256) {
    int dI = j >> 6, c = j & 63;
    kv_out[((size_t)bh * kDD + dI) * 64 + c] = KV[c * kDP + dI];
  }
  for (int dI = tid; dI < kDD; dI += 256)
    kst_out[(size_t)bh * kDD + dI] = kst[dI];
}

// ---------------------------------------------------------------------------
// Window attention: all L queries attend to the same shifted 64-entry buffer.
// One block per (b,h); K and V^T tiles cached in LDS as bf16.
// ---------------------------------------------------------------------------
__global__ __launch_bounds__(256) void window_attn(
    const float* __restrict__ q2, const float* __restrict__ k2,
    const float* __restrict__ v2, const float* __restrict__ kbuf,
    const float* __restrict__ vbuf, float* __restrict__ win)
{
  __shared__ bf16 ukb[64 * 64];       // uk[j][c]
  __shared__ bf16 uvT[64 * 64];       // uv^T[c][j]
  __shared__ bf16 attb[8 * 16 * 64];  // per-wave attn rows

  const int tid  = threadIdx.x;
  const int wv   = tid >> 5;
  const int lane = tid & 31;
  const int r    = lane & 15;
  const int hf   = lane >> 4;
  const int bb   = blockIdx.x >> 4;
  const int hh   = blockIdx.x & 15;
  const v8f vz = {0.f,0.f,0.f,0.f,0.f,0.f,0.f,0.f};

  for (int j = tid; j < 4096; j += 256) {
    int jj = j >> 6, cc = j & 63;
    float kv, vvv;
    if (jj < 63) {
      kv  = kbuf[((size_t)(hh * 64) + jj + 1) * 64 + cc];
      vvv = vbuf[((size_t)(hh * 64) + jj + 1) * 64 + cc];
    } else {
      kv  = k2[((size_t)(bb * kL + kL - 1)) * kD + hh * 64 + cc];
      vvv = v2[((size_t)(bb * kL + kL - 1)) * kD + hh * 64 + cc];
    }
    ukb[jj * 64 + cc] = (bf16)kv;
    uvT[cc * 64 + jj] = (bf16)vvv;
  }
  __syncthreads();

  for (int it = 0; it < kL / 128; ++it) {
    int row0 = it * 128 + wv * 16;
    v8f sc[4] = {vz, vz, vz, vz};
    const float* qrow = q2 + ((size_t)(bb * kL + row0 + r)) * kD + hh * 64;
    for (int kb = 0; kb < 2; ++kb) {
      v16bf a;
#pragma unroll
      for (int e = 0; e < 8; ++e) {
        a[e]     = (bf16)(0.125f * qrow[kb * 32 + 16 * hf + e]);
        a[8 + e] = (bf16)(0.125f * qrow[kb * 32 + 16 * hf + 8 + e]);
      }
#pragma unroll
      for (int nt = 0; nt < 4; ++nt) {
        v16bf b;
        int n = nt * 16 + r;
#pragma unroll
        for (int e = 0; e < 16; ++e) b[e] = ukb[n * 64 + kb * 32 + 16 * hf + e];
        sc[nt] = wmma_bf16(a, b, sc[nt]);
      }
    }
    // softmax over 64 cols: per row reduce across the 16-lane half
    bf16* aw = attb + wv * 1024;
#pragma unroll
    for (int r8 = 0; r8 < 8; ++r8) {
      float m = fmaxf(fmaxf(sc[0][r8], sc[1][r8]), fmaxf(sc[2][r8], sc[3][r8]));
      for (int off = 1; off < 16; off <<= 1) m = fmaxf(m, __shfl_xor(m, off, 32));
      float ex[4], s = 0.f;
#pragma unroll
      for (int nt = 0; nt < 4; ++nt) { ex[nt] = __expf(sc[nt][r8] - m); s += ex[nt]; }
      for (int off = 1; off < 16; off <<= 1) s += __shfl_xor(s, off, 32);
      float inv = 1.f / s;
#pragma unroll
      for (int nt = 0; nt < 4; ++nt)
        aw[(r8 + 8 * hf) * 64 + nt * 16 + r] = (bf16)(ex[nt] * inv);
    }
    __syncthreads();
    v8f wacc[4] = {vz, vz, vz, vz};
    for (int kb = 0; kb < 2; ++kb) {
      v16bf a;
#pragma unroll
      for (int e = 0; e < 8; ++e) {
        a[e]     = aw[r * 64 + kb * 32 + 16 * hf + e];
        a[8 + e] = aw[r * 64 + kb * 32 + 16 * hf + 8 + e];
      }
#pragma unroll
      for (int nt = 0; nt < 4; ++nt) {
        v16bf b;
        int n = nt * 16 + r;
#pragma unroll
        for (int e = 0; e < 16; ++e) b[e] = uvT[n * 64 + kb * 32 + 16 * hf + e];
        wacc[nt] = wmma_bf16(a, b, wacc[nt]);
      }
    }
#pragma unroll
    for (int nt = 0; nt < 4; ++nt) {
      int col = hh * 64 + nt * 16 + r;
#pragma unroll
      for (int r8 = 0; r8 < 8; ++r8)
        win[((size_t)(bb * kL + row0 + r8 + 8 * hf)) * kD + col] = wacc[nt][r8];
    }
    __syncthreads();
  }
}

// ---------------------------------------------------------------------------
// out = LayerNorm(xa + xb) * gamma + beta ; one block per row (d=1024)
// ---------------------------------------------------------------------------
__global__ __launch_bounds__(256) void ln_residual(
    const float* __restrict__ xa, const float* __restrict__ xb,
    const float* __restrict__ gamma, const float* __restrict__ beta,
    float* __restrict__ out)
{
  __shared__ float ssum[256], ssq[256];
  const int tid = threadIdx.x;
  const size_t base = (size_t)blockIdx.x * kD;
  float vals[4], s = 0.f, q = 0.f;
#pragma unroll
  for (int i = 0; i < 4; ++i) {
    int j = tid + i * 256;
    float v = xa[base + j] + xb[base + j];
    vals[i] = v; s += v; q += v * v;
  }
  ssum[tid] = s; ssq[tid] = q;
  __syncthreads();
  for (int off = 128; off > 0; off >>= 1) {
    if (tid < off) { ssum[tid] += ssum[tid + off]; ssq[tid] += ssq[tid + off]; }
    __syncthreads();
  }
  float mu  = ssum[0] * (1.f / kD);
  float var = ssq[0] * (1.f / kD) - mu * mu;
  float inv = rsqrtf(var + 1e-5f);
#pragma unroll
  for (int i = 0; i < 4; ++i) {
    int j = tid + i * 256;
    out[base + j] = (vals[i] - mu) * inv * gamma[j] + beta[j];
  }
}

// ---------------------------------------------------------------------------
extern "C" void kernel_launch(void* const* d_in, const int* in_sizes, int n_in,
                              void* d_out, int out_size, void* d_ws, size_t ws_size,
                              hipStream_t stream) {
  const float* x      = (const float*)d_in[0];
  const float* kv0    = (const float*)d_in[1];
  const float* kst0   = (const float*)d_in[2];
  const float* kbuf   = (const float*)d_in[3];
  const float* vbuf   = (const float*)d_in[4];
  const float* Wq_lin = (const float*)d_in[5];
  const float* bq_lin = (const float*)d_in[6];
  const float* Wk_lin = (const float*)d_in[7];
  const float* bk_lin = (const float*)d_in[8];
  const float* Wv_lin = (const float*)d_in[9];
  const float* bv_lin = (const float*)d_in[10];
  const float* Wo_lin = (const float*)d_in[11];
  const float* bo_lin = (const float*)d_in[12];
  const float* Wq     = (const float*)d_in[13];
  const float* bq     = (const float*)d_in[14];
  const float* Wk     = (const float*)d_in[15];
  const float* bk     = (const float*)d_in[16];
  const float* Wv     = (const float*)d_in[17];
  const float* bv     = (const float*)d_in[18];
  const float* Wo     = (const float*)d_in[19];
  const float* bo     = (const float*)d_in[20];
  const float* gamma  = (const float*)d_in[21];
  const float* beta   = (const float*)d_in[22];

  float* out     = (float*)d_out;
  float* kv_new  = out + (size_t)kB * kL * kD;               // (B,h,D,hd)
  float* kst_new = kv_new + (size_t)kB * kH * kDD * kHD;     // (B,h,D)

  const size_t R = (size_t)kB * kL * kD;   // 8,388,608 floats per region
  float* r0 = (float*)d_ws;
  float* r1 = r0 + R;
  float* r2 = r1 + R;
  float* r3 = r2 + R;
  float* r4 = r3 + R;
  float* qlin = r0;                                   // (B,L,h*f)
  float* klin = r0 + (size_t)kB * kL * kH * kF;       // (B,L,h*f)

  const int M = kB * kL;
  dim3 blk(256);

  gemm_bias_wmma<<<dim3(2, M / 128), blk, 0, stream>>>(x, Wq_lin, bq_lin, qlin, M, 256, kD);
  gemm_bias_wmma<<<dim3(2, M / 128), blk, 0, stream>>>(x, Wk_lin, bk_lin, klin, M, 256, kD);
  gemm_bias_wmma<<<dim3(8, M / 128), blk, 0, stream>>>(x, Wv_lin, bv_lin, r1, M, kD, kD);

  linear_scan_wmma<<<dim3(kB * kH), blk, SCAN_LDS_SIZE, stream>>>(
      qlin, klin, r1, kv0, kst0, r2, kv_new, kst_new);

  gemm_bias_wmma<<<dim3(8, M / 128), blk, 0, stream>>>(r2, Wo_lin, bo_lin, r1, M, kD, kD);
  ln_residual<<<dim3(M), blk, 0, stream>>>(x, r1, gamma, beta, r0);  // x1 -> r0

  gemm_bias_wmma<<<dim3(8, M / 128), blk, 0, stream>>>(r0, Wq, bq, r1, M, kD, kD);  // q2
  gemm_bias_wmma<<<dim3(8, M / 128), blk, 0, stream>>>(r0, Wk, bk, r2, M, kD, kD);  // k2
  gemm_bias_wmma<<<dim3(8, M / 128), blk, 0, stream>>>(r0, Wv, bv, r3, M, kD, kD);  // v2

  window_attn<<<dim3(kB * kH), blk, 0, stream>>>(r1, r2, r3, kbuf, vbuf, r4);

  gemm_bias_wmma<<<dim3(8, M / 128), blk, 0, stream>>>(r4, Wo, bo, r2, M, kD, kD);
  ln_residual<<<dim3(M), blk, 0, stream>>>(r0, r2, gamma, beta, out);
}
